// GNNModule_21844203667553
// MI455X (gfx1250) — compile-verified, matrix operand order checked by reading, not run on hardware
//
#include <hip/hip_runtime.h>
#include <hip/hip_bf16.h>

typedef __attribute__((ext_vector_type(16))) __bf16 v16bf;
typedef __attribute__((ext_vector_type(8)))  float  v8f;

#define NNODES 100000
#define NEDGES 1600000
#define DIN    64
#define DHID   128
#define NGRAPH 64
#define BN_EPS 1e-5f

// ---------------- utility kernels ----------------

__global__ void k_zero(float* __restrict__ p, int n) {
    int i = blockIdx.x * blockDim.x + threadIdx.x;
    if (i < n) p[i] = 0.0f;
}

__global__ void k_degree(const int* __restrict__ dst, float* __restrict__ deg, int E) {
    int e = blockIdx.x * blockDim.x + threadIdx.x;
    if (e < E) atomicAdd(&deg[dst[e]], 1.0f);
}

// Pack W [KDIM x 128 fp32, row-major] into per-lane WMMA B fragments (bf16).
// Fragment index: ((ntile * (KDIM/32) + ks) * 32 + lane) -> 16 contiguous bf16,
// matching the 16x16x32 bf16 B layout: lanes 0-15 hold K {ks*32+0..7, +16..23},
// lanes 16-31 hold K {+8..15, +24..31}, col = ntile*16 + (lane&15).
template<int KDIM>
__global__ void k_pack_B(const float* __restrict__ W, __bf16* __restrict__ P) {
    const int NK = KDIM / 32;
    int t = blockIdx.x * blockDim.x + threadIdx.x;
    if (t >= 8 * NK * 32) return;
    int lane  = t & 31;
    int ks    = (t >> 5) % NK;
    int ntile = (t >> 5) / NK;
    int col   = ntile * 16 + (lane & 15);
    int khalf = (lane >> 4) << 3;
    __bf16* o = P + (size_t)t * 16;
#pragma unroll
    for (int j = 0; j < 8; ++j) {
        o[j]     = (__bf16)W[(size_t)(ks * 32 + khalf + j)      * DHID + col];
        o[8 + j] = (__bf16)W[(size_t)(ks * 32 + khalf + 16 + j) * DHID + col];
    }
}

// Edge-parallel scatter-add of h[src] into agg[dst]. agg (51 MB) is L2-resident
// on MI455X (192 MB L2), so these float atomics resolve on-chip.
template<int D>
__global__ void k_scatter(const float* __restrict__ h, const int* __restrict__ src,
                          const int* __restrict__ dst, float* __restrict__ agg, int E) {
    const int CH = D / 4;
    long long t = (long long)blockIdx.x * blockDim.x + threadIdx.x;
    if (t >= (long long)E * CH) return;
    int e = (int)(t / CH);
    int c = (int)(t % CH);
    int s = src[e], d = dst[e];
    float4 v = ((const float4*)(h + (size_t)s * D))[c];
    float* out = agg + (size_t)d * D + c * 4;
    atomicAdd(out + 0, v.x);
    atomicAdd(out + 1, v.y);
    atomicAdd(out + 2, v.z);
    atomicAdd(out + 3, v.w);
}

// z = h + agg / max(deg,1), written in place over agg
template<int D>
__global__ void k_combine(const float* __restrict__ h, const float* __restrict__ deg,
                          float* __restrict__ agg, int n) {
    long long t = (long long)blockIdx.x * blockDim.x + threadIdx.x;
    if (t >= (long long)n * D) return;
    int node = (int)(t / D);
    float inv = 1.0f / fmaxf(deg[node], 1.0f);
    agg[t] = h[t] + agg[t] * inv;
}

// ---------------- fused WMMA GEMM + bias + BN + relu ----------------
// A: [100000 x KDIM] fp32   Wp: pre-packed B fragments (bf16)
// out[row, col] = EPI==0 ? relu(bn(x + bias))         (inner MLP lin1)
//                EPI==1 ? relu(bn(relu(x + bias)))    (lin2 + GIN act + outer BN)
// One 16x16 tile per wave, 8 waves cover 128 cols. A tile staged once per
// block via async global->LDS copy (ASYNCcnt), fragments read with ds_load.
template<int KDIM, int EPI>
__global__ void __launch_bounds__(256)
k_gemm_bn_relu(const float* __restrict__ A,
               const __bf16* __restrict__ Wp,
               const float* __restrict__ bias,
               const float* __restrict__ gamma, const float* __restrict__ beta,
               const float* __restrict__ mean,  const float* __restrict__ var,
               float* __restrict__ out) {
    const int NK  = KDIM / 32;
    const int PAD = 4;                         // +4 floats/row: conflict-free ds_load_b128
    __shared__ float sA[16][KDIM + PAD];

    const int tid   = threadIdx.x;
    const int lane  = tid & 31;
    const int wave  = tid >> 5;                // N-tile 0..7
    const int mbase = blockIdx.x * 16;         // 100000 = 6250*16, no tail

    // ---- cooperative async copy: 16 x KDIM fp32 tile -> LDS ----
    const int NV = (16 * KDIM) / 4;            // float4 chunks (256 or 512)
    const float4* gsrc = (const float4*)(A + (size_t)mbase * KDIM);
#pragma unroll
    for (int i = tid; i < NV; i += 256) {
        int row  = i / (KDIM / 4);
        int col4 = i % (KDIM / 4);
        uint32_t ldsoff = (uint32_t)(size_t)(&sA[row][col4 * 4]);
        uint64_t gaddr  = (uint64_t)(size_t)(gsrc + i);
        asm volatile("global_load_async_to_lds_b128 %0, %1, off"
                     :: "v"(ldsoff), "v"(gaddr) : "memory");
    }
    asm volatile("s_wait_asynccnt 0" ::: "memory");
    __syncthreads();

    const int row   = lane & 15;
    const int khalf = (lane >> 4) << 3;        // 0 for lanes 0-15, 8 for 16-31
    const int col   = wave * 16 + row;
    const v16bf* Bf = (const v16bf*)Wp;

    v8f acc = {};
#pragma unroll
    for (int kb32 = 0; kb32 < NK; ++kb32) {
        const int kb = kb32 * 32;
        // A fragment from LDS: lane holds K {kb+khalf..+7} U {kb+khalf+16..+23}
        float4 a0 = *(const float4*)(&sA[row][kb + khalf + 0]);
        float4 a1 = *(const float4*)(&sA[row][kb + khalf + 4]);
        float4 a2 = *(const float4*)(&sA[row][kb + khalf + 16]);
        float4 a3 = *(const float4*)(&sA[row][kb + khalf + 20]);
        v16bf a;
        a[0]  = (__bf16)a0.x; a[1]  = (__bf16)a0.y; a[2]  = (__bf16)a0.z; a[3]  = (__bf16)a0.w;
        a[4]  = (__bf16)a1.x; a[5]  = (__bf16)a1.y; a[6]  = (__bf16)a1.z; a[7]  = (__bf16)a1.w;
        a[8]  = (__bf16)a2.x; a[9]  = (__bf16)a2.y; a[10] = (__bf16)a2.z; a[11] = (__bf16)a2.w;
        a[12] = (__bf16)a3.x; a[13] = (__bf16)a3.y; a[14] = (__bf16)a3.z; a[15] = (__bf16)a3.w;

        // B fragment: one 32-byte contiguous load (pre-packed)
        v16bf b = Bf[(size_t)(wave * NK + kb32) * 32 + lane];

        acc = __builtin_amdgcn_wmma_f32_16x16x32_bf16(
            /*neg_a=*/false, a, /*neg_b=*/false, b,
            /*c_mod=*/(short)0, acc, /*reuse_a=*/false, /*reuse_b=*/false);
    }

    // Epilogue from C/D layout: VGPR r -> row mbase+khalf+r, N = col
    float sc = gamma[col] * rsqrtf(var[col] + BN_EPS);
    float sh = beta[col] - mean[col] * sc;
    float bi = bias[col];
    int row0 = mbase + khalf;
#pragma unroll
    for (int r = 0; r < 8; ++r) {
        float x = acc[r] + bi;
        if (EPI == 1) x = fmaxf(x, 0.0f);
        x = fmaxf(x * sc + sh, 0.0f);
        out[(size_t)(row0 + r) * DHID + col] = x;
    }
}

// ---------------- readout ----------------

__global__ void k_pool_accum(const float* __restrict__ h, const int* __restrict__ n2g,
                             float* __restrict__ hg, float* __restrict__ cnt) {
    long long t = (long long)blockIdx.x * blockDim.x + threadIdx.x;
    if (t >= (long long)NNODES * DHID) return;
    int node = (int)(t >> 7);
    int d    = (int)(t & 127);
    int g    = n2g[node];
    atomicAdd(&hg[(size_t)g * DHID + d], h[t]);
    if (d == 0) atomicAdd(&cnt[g], 1.0f);
}

__global__ void k_pool_final(const float* __restrict__ hg, const float* __restrict__ cnt,
                             float* __restrict__ out) {
    int t = blockIdx.x * blockDim.x + threadIdx.x;
    if (t >= NGRAPH * DHID) return;
    out[t] = hg[t] / fmaxf(cnt[t >> 7], 1.0f);
}

// ---------------- driver ----------------

static inline int nblk(long long n) { return (int)((n + 255) / 256); }

extern "C" void kernel_launch(void* const* d_in, const int* in_sizes, int n_in,
                              void* d_out, int out_size, void* d_ws, size_t ws_size,
                              hipStream_t stream) {
    // Input order: h, mlp1{w1,b1,bn{g,b,m,v},w2,b2}, bn1{g,b,m,v},
    //              mlp2{w1,b1,bn{g,b,m,v},w2,b2}, bn2{g,b,m,v}, src, dst, n2g, G
    const float* h      = (const float*)d_in[0];
    const float* w1a    = (const float*)d_in[1];
    const float* b1a    = (const float*)d_in[2];
    const float* bn1a_g = (const float*)d_in[3];
    const float* bn1a_b = (const float*)d_in[4];
    const float* bn1a_m = (const float*)d_in[5];
    const float* bn1a_v = (const float*)d_in[6];
    const float* w2a    = (const float*)d_in[7];
    const float* b2a    = (const float*)d_in[8];
    const float* bn1_g  = (const float*)d_in[9];
    const float* bn1_b  = (const float*)d_in[10];
    const float* bn1_m  = (const float*)d_in[11];
    const float* bn1_v  = (const float*)d_in[12];
    const float* w1b    = (const float*)d_in[13];
    const float* b1b    = (const float*)d_in[14];
    const float* bn2a_g = (const float*)d_in[15];
    const float* bn2a_b = (const float*)d_in[16];
    const float* bn2a_m = (const float*)d_in[17];
    const float* bn2a_v = (const float*)d_in[18];
    const float* w2b    = (const float*)d_in[19];
    const float* b2b    = (const float*)d_in[20];
    const float* bn2_g  = (const float*)d_in[21];
    const float* bn2_b  = (const float*)d_in[22];
    const float* bn2_m  = (const float*)d_in[23];
    const float* bn2_v  = (const float*)d_in[24];
    const int* src = (const int*)d_in[25];
    const int* dst = (const int*)d_in[26];
    const int* n2g = (const int*)d_in[27];
    float* out = (float*)d_out;

    // Workspace carve-up (float region first; all offsets keep 32B alignment)
    float* deg  = (float*)d_ws;                               // N
    float* bufA = deg  + NNODES;                              // N*128 (agg/z, then h2)
    float* bufT = bufA + (size_t)NNODES * DHID;               // N*128 (hidden)
    float* bufH = bufT + (size_t)NNODES * DHID;               // N*128 (layer1 out)
    float* hg   = bufH + (size_t)NNODES * DHID;               // G*128
    float* cnt  = hg   + NGRAPH * DHID;                       // G
    __bf16* wb1a = (__bf16*)(cnt + NGRAPH);                   // packed frags, 64*128
    __bf16* wb2a = wb1a + DIN  * DHID;                        // 128*128
    __bf16* wb1b = wb2a + DHID * DHID;
    __bf16* wb2b = wb1b + DHID * DHID;

    // Pre-pack weights into WMMA B-fragment layout (tiny)
    k_pack_B<DIN> <<<nblk(8 * (DIN  / 32) * 32), 256, 0, stream>>>(w1a, wb1a);
    k_pack_B<DHID><<<nblk(8 * (DHID / 32) * 32), 256, 0, stream>>>(w2a, wb2a);
    k_pack_B<DHID><<<nblk(8 * (DHID / 32) * 32), 256, 0, stream>>>(w1b, wb1b);
    k_pack_B<DHID><<<nblk(8 * (DHID / 32) * 32), 256, 0, stream>>>(w2b, wb2b);

    // Degrees (shared by both layers)
    k_zero<<<nblk(NNODES), 256, 0, stream>>>(deg, NNODES);
    k_degree<<<nblk(NEDGES), 256, 0, stream>>>(dst, deg, NEDGES);

    const int gemmGrid = NNODES / 16;   // 6250

    // ---- Layer 1 (D=64 in) ----
    k_zero<<<nblk((long long)NNODES * DIN), 256, 0, stream>>>(bufA, NNODES * DIN);
    k_scatter<DIN><<<nblk((long long)NEDGES * (DIN / 4)), 256, 0, stream>>>(h, src, dst, bufA, NEDGES);
    k_combine<DIN><<<nblk((long long)NNODES * DIN), 256, 0, stream>>>(h, deg, bufA, NNODES);
    k_gemm_bn_relu<DIN, 0><<<gemmGrid, 256, 0, stream>>>(bufA, wb1a, b1a,
        bn1a_g, bn1a_b, bn1a_m, bn1a_v, bufT);
    k_gemm_bn_relu<DHID, 1><<<gemmGrid, 256, 0, stream>>>(bufT, wb2a, b2a,
        bn1_g, bn1_b, bn1_m, bn1_v, bufH);

    // ---- Layer 2 (D=128 in) ----
    k_zero<<<nblk((long long)NNODES * DHID), 256, 0, stream>>>(bufA, NNODES * DHID);
    k_scatter<DHID><<<nblk((long long)NEDGES * (DHID / 4)), 256, 0, stream>>>(bufH, src, dst, bufA, NEDGES);
    k_combine<DHID><<<nblk((long long)NNODES * DHID), 256, 0, stream>>>(bufH, deg, bufA, NNODES);
    k_gemm_bn_relu<DHID, 0><<<gemmGrid, 256, 0, stream>>>(bufA, wb1b, b1b,
        bn2a_g, bn2a_b, bn2a_m, bn2a_v, bufT);
    k_gemm_bn_relu<DHID, 1><<<gemmGrid, 256, 0, stream>>>(bufT, wb2b, b2b,
        bn2_g, bn2_b, bn2_m, bn2_v, bufA);   // layer2 output into bufA

    // ---- Mean readout ----
    k_zero<<<nblk(NGRAPH * DHID + NGRAPH), 256, 0, stream>>>(hg, NGRAPH * DHID + NGRAPH);
    k_pool_accum<<<nblk((long long)NNODES * DHID), 256, 0, stream>>>(bufA, n2g, hg, cnt);
    k_pool_final<<<nblk(NGRAPH * DHID), 256, 0, stream>>>(hg, cnt, out);
}
